// MaskedAutoencoderViT_29008209117189
// MI455X (gfx1250) — matrix operand order repaired and checked by reading, not compiled, not run on hardware
//
#include <hip/hip_runtime.h>
#include <hip/hip_bf16.h>
#include <math.h>

// ---------------------------------------------------------------------------
// MAE-ViT forward for MI455X (gfx1250): WMMA f16 GEMMs + VALU attention.
// GEMM: 128x128x32 tiles, double-buffered LDS, straight-line staging,
// global_prefetch for tile k+2, v_wmma_f32_16x16x32_f16 core.
// ---------------------------------------------------------------------------

typedef __attribute__((ext_vector_type(16))) _Float16 v16h;
typedef __attribute__((ext_vector_type(8)))  _Float16 v8h;
typedef __attribute__((ext_vector_type(8)))  float    v8f;

#define BATCH     8
#define SEQL      2000
#define DIM       512
#define HD        32
#define LEN_KEEP  500
#define ENC_N     501
#define DEC_N     2001
#define ENC_DEPTH 24
#define DEC_DEPTH 8
#define WSROWS    16128          // == max grid.y * BM; all GEMM A-operands live here

#define BM 128
#define BN 128
#define BK 32

// ------------------------------- WMMA GEMM --------------------------------
// C[M,N] = act( A[M,K] @ W[K,N] + bias[N] (+ res[M,N]) )
// A must be a workspace buffer with >= gridDim.y*BM rows (loads unguarded).
__global__ __launch_bounds__(256) void gemm_wmma(
    const float* __restrict__ A, const float* __restrict__ W,
    const float* __restrict__ bias, const float* __restrict__ res,
    float* __restrict__ C, int M, int N, int K, int use_gelu)
{
  __shared__ __align__(16) _Float16 As[2][BM][BK];   // 2 x 8 KB
  __shared__ __align__(16) _Float16 Bs[2][BN][BK];   // 2 x 8 KB, stored [n][k]
  const int tid  = threadIdx.x;
  const int m0   = blockIdx.y * BM;
  const int n0   = blockIdx.x * BN;
  const int wave = tid >> 5;
  const int lane = tid & 31;
  const int wm   = wave >> 2;       // 0..1  -> 64-row slab
  const int wn   = wave & 3;        // 0..3  -> 32-col slab
  const int half = lane >> 4;
  const int l16  = lane & 15;

  // per-thread staging coordinates (4 float4 of A, 4 float4 of W per step)
  int rA[4], cA[4], rW[4], cW[4];
#pragma unroll
  for (int it = 0; it < 4; ++it) {
    int e = tid + it * 256;           // 0..1023 float4 slots
    rA[it] = e >> 3;  cA[it] = (e & 7)  * 4;   // A: 128 rows x 8 float4
    rW[it] = e >> 5;  cW[it] = (e & 31) * 4;   // W: 32 rows x 32 float4
  }

  v8f acc[4][2];
#pragma unroll
  for (int mi = 0; mi < 4; ++mi)
#pragma unroll
    for (int ni = 0; ni < 2; ++ni)
#pragma unroll
      for (int i = 0; i < 8; ++i) acc[mi][ni][i] = 0.f;

  // ---- prologue: load + stage k0 = 0 ----
  float4 av[4], wv[4];
#pragma unroll
  for (int it = 0; it < 4; ++it) {
    av[it] = *(const float4*)(A + (size_t)(m0 + rA[it]) * K + cA[it]);
    wv[it] = *(const float4*)(W + (size_t)rW[it] * N + n0 + cW[it]);
  }
#pragma unroll
  for (int it = 0; it < 4; ++it) {
    As[0][rA[it]][cA[it]+0] = (_Float16)av[it].x;
    As[0][rA[it]][cA[it]+1] = (_Float16)av[it].y;
    As[0][rA[it]][cA[it]+2] = (_Float16)av[it].z;
    As[0][rA[it]][cA[it]+3] = (_Float16)av[it].w;
    Bs[0][cW[it]+0][rW[it]] = (_Float16)wv[it].x;
    Bs[0][cW[it]+1][rW[it]] = (_Float16)wv[it].y;
    Bs[0][cW[it]+2][rW[it]] = (_Float16)wv[it].z;
    Bs[0][cW[it]+3][rW[it]] = (_Float16)wv[it].w;
  }
  __syncthreads();

  int buf = 0;
  for (int k0 = 0; k0 < K; k0 += BK) {
    const bool has_next = (k0 + BK) < K;   // uniform
    if (has_next) {
      // issue next tile's global loads before computing (latency overlap)
#pragma unroll
      for (int it = 0; it < 4; ++it) {
        av[it] = *(const float4*)(A + (size_t)(m0 + rA[it]) * K + (k0 + BK) + cA[it]);
        wv[it] = *(const float4*)(W + (size_t)(k0 + BK + rW[it]) * N + n0 + cW[it]);
      }
      if (k0 + 2 * BK < K) {
        __builtin_prefetch(A + (size_t)(m0 + rA[0]) * K + (k0 + 2 * BK) + cA[0], 0, 1);
        __builtin_prefetch(W + (size_t)(k0 + 2 * BK + rW[0]) * N + n0 + cW[0], 0, 1);
      }
    }

    // ---- fragments per CDNA5 16-bit A/B layouts ----
    v16h af[4];
#pragma unroll
    for (int mi = 0; mi < 4; ++mi) {
      int row = wm * 64 + mi * 16 + l16;
      const v8h* pa = (const v8h*)(&As[buf][row][0]);
      v8h lo = pa[half];          // e=0..7  -> K = e + 8*half
      v8h hi = pa[2 + half];      // e=8..15 -> K = e + 8 + 8*half
      v16h a;
#pragma unroll
      for (int i = 0; i < 8; ++i) { a[i] = lo[i]; a[8 + i] = hi[i]; }
      af[mi] = a;
    }
    v16h bf[2];
#pragma unroll
    for (int ni = 0; ni < 2; ++ni) {
      int col = wn * 32 + ni * 16 + l16;
      const v8h* pb = (const v8h*)(&Bs[buf][col][0]);
      v8h lo = pb[half];          // K = 8*half + 0..7
      v8h hi = pb[2 + half];      // K = 16 + 8*half + 0..7
      v16h b;
#pragma unroll
      for (int i = 0; i < 8; ++i) { b[i] = lo[i]; b[8 + i] = hi[i]; }
      bf[ni] = b;
    }
#pragma unroll
    for (int mi = 0; mi < 4; ++mi)
#pragma unroll
      for (int ni = 0; ni < 2; ++ni)
        acc[mi][ni] = __builtin_amdgcn_wmma_f32_16x16x32_f16(
            false, af[mi], false, bf[ni], (short)0, acc[mi][ni], false, false);

    if (has_next) {
      const int nb = buf ^ 1;
#pragma unroll
      for (int it = 0; it < 4; ++it) {
        As[nb][rA[it]][cA[it]+0] = (_Float16)av[it].x;
        As[nb][rA[it]][cA[it]+1] = (_Float16)av[it].y;
        As[nb][rA[it]][cA[it]+2] = (_Float16)av[it].z;
        As[nb][rA[it]][cA[it]+3] = (_Float16)av[it].w;
        Bs[nb][cW[it]+0][rW[it]] = (_Float16)wv[it].x;
        Bs[nb][cW[it]+1][rW[it]] = (_Float16)wv[it].y;
        Bs[nb][cW[it]+2][rW[it]] = (_Float16)wv[it].z;
        Bs[nb][cW[it]+3][rW[it]] = (_Float16)wv[it].w;
      }
      __syncthreads();
      buf = nb;
    }
  }

  // ---- epilogue: bias / residual / exact GELU ----
#pragma unroll
  for (int mi = 0; mi < 4; ++mi)
#pragma unroll
    for (int ni = 0; ni < 2; ++ni) {
      int col = n0 + wn * 32 + ni * 16 + l16;
      if (col >= N) continue;
      float bv = bias ? bias[col] : 0.f;
#pragma unroll
      for (int i = 0; i < 8; ++i) {
        int row = m0 + wm * 64 + mi * 16 + half * 8 + i;
        if (row < M) {
          float v = acc[mi][ni][i] + bv;
          if (res) v += res[(size_t)row * N + col];
          if (use_gelu) v = 0.5f * v * (1.f + erff(v * 0.70710678118654752f));
          C[(size_t)row * N + col] = v;
        }
      }
    }
}

// ------------------------------- LayerNorm --------------------------------
__global__ __launch_bounds__(256) void layernorm_k(
    const float* __restrict__ x, const float* __restrict__ w,
    const float* __restrict__ b, float* __restrict__ y, int rows)
{
  int wave = threadIdx.x >> 5, lane = threadIdx.x & 31;
  int row = blockIdx.x * 8 + wave;
  if (row >= rows) return;
  const float* xr = x + (size_t)row * DIM;
  float s = 0.f, ss = 0.f;
  for (int c = lane; c < DIM; c += 32) { float v = xr[c]; s += v; ss += v * v; }
  for (int off = 16; off; off >>= 1) {
    s  += __shfl_xor(s,  off, 32);
    ss += __shfl_xor(ss, off, 32);
  }
  float mean = s * (1.f / DIM);
  float var  = ss * (1.f / DIM) - mean * mean;
  float r    = rsqrtf(var + 1e-5f);
  float* yr  = y + (size_t)row * DIM;
  for (int c = lane; c < DIM; c += 32) yr[c] = (xr[c] - mean) * r * w[c] + b[c];
}

// -------------------------- argsort / mask --------------------------------
__global__ void argsort_k(const float* __restrict__ noise, int* __restrict__ ids_shuffle,
                          int* __restrict__ ids_restore, float* __restrict__ mask_out)
{
  int b = blockIdx.x;
  __shared__ float sn[2048];
  for (int i = threadIdx.x; i < SEQL; i += blockDim.x) sn[i] = noise[b * SEQL + i];
  __syncthreads();
  for (int i = threadIdx.x; i < SEQL; i += blockDim.x) {
    float v = sn[i];
    int r = 0;
    for (int j = 0; j < SEQL; ++j) {
      float u = sn[j];
      r += (u < v) || (u == v && j < i);     // stable argsort rank
    }
    ids_restore[b * SEQL + i] = r;
    ids_shuffle[b * SEQL + r] = i;
    mask_out[b * SEQL + i] = (r >= LEN_KEEP) ? 1.f : 0.f;
  }
}

// ---------------------------- encoder embed -------------------------------
__global__ void enc_embed_k(const float* __restrict__ expr, const int* __restrict__ idx,
                            const int* __restrict__ ids_shuffle,
                            const float* __restrict__ pos_embed,
                            const float* __restrict__ cls_token,
                            const float* __restrict__ ew, const float* __restrict__ eb,
                            float* __restrict__ x)
{
  int row = blockIdx.x;            // b*(ENC_N)+t
  int b = row / ENC_N, t = row % ENC_N;
  float* xr = x + (size_t)row * DIM;
  if (t == 0) {
    for (int c = threadIdx.x; c < DIM; c += blockDim.x) xr[c] = cls_token[c];
  } else {
    int p  = ids_shuffle[b * SEQL + (t - 1)];
    float e = expr[b * SEQL + p];
    const float* pe = pos_embed + (size_t)idx[b * SEQL + p] * DIM;
    for (int c = threadIdx.x; c < DIM; c += blockDim.x)
      xr[c] = e * ew[c] + eb[c] + pe[c];
  }
}

// ------------------------- local attention (head 0) -----------------------
__global__ __launch_bounds__(256) void local_attn_k(
    const float* __restrict__ Q, const float* __restrict__ KV,
    const float* __restrict__ kvb, float* __restrict__ Out,
    int nt, int npad, int padlen)
{
  int win = blockIdx.x, b = blockIdx.y;
  __shared__ float kvS[192][32];
  __shared__ float maskS[192];
  size_t base = (size_t)b * nt;
  for (int e = threadIdx.x; e < 192 * 32; e += blockDim.x) {
    int j = e >> 5, d = e & 31;
    int kpos = (win - 1) * 64 + j;
    float v = 0.f;
    if (kpos >= 0 && kpos < npad) {
      int t = kpos - padlen;
      v = (t < 0) ? kvb[d] : KV[(base + t) * DIM + d];
    }
    kvS[j][d] = v;
    if (d == 0) maskS[j] = (kpos >= 0 && kpos < npad) ? 0.f : -1e9f;
  }
  __syncthreads();

  int tq = threadIdx.x;
  if (tq < 64) {
    int qpos = win * 64 + tq;
    int t = qpos - padlen;
    if (t >= 0 && t < nt) {
      float qv[32];
      const float* qr = Q + (base + t) * DIM;
#pragma unroll
      for (int d = 0; d < 32; ++d) qv[d] = qr[d];
      const float scale = 0.17677669529663687f;  // 32^-0.5
      float m = -1e30f;
      for (int j = 0; j < 192; ++j) {
        float s = 0.f;
#pragma unroll
        for (int d = 0; d < 32; ++d) s += qv[d] * kvS[j][d];
        s = s * scale + maskS[j];
        m = fmaxf(m, s);
      }
      float denom = 0.f;
      float accv[32];
#pragma unroll
      for (int d = 0; d < 32; ++d) accv[d] = 0.f;
      for (int j = 0; j < 192; ++j) {
        float s = 0.f;
#pragma unroll
        for (int d = 0; d < 32; ++d) s += qv[d] * kvS[j][d];
        s = s * scale + maskS[j];
        float p = expf(s - m);
        denom += p;
#pragma unroll
        for (int d = 0; d < 32; ++d) accv[d] += p * kvS[j][d];
      }
      float inv = 1.f / denom;
      float* orow = Out + (base + t) * DIM;
#pragma unroll
      for (int d = 0; d < 32; ++d) orow[d] = accv[d] * inv;
    }
  }
}

// ------------------ linear attention (heads 1..15) ------------------------
__global__ void qsoftmax_k(float* __restrict__ Q, int rows)
{
  int g = blockIdx.x * blockDim.x + threadIdx.x;
  if (g >= rows * 15) return;
  int row = g / 15, h = g % 15 + 1;
  float* q = Q + (size_t)row * DIM + h * HD;
  float m = -1e30f;
#pragma unroll
  for (int d = 0; d < 32; ++d) m = fmaxf(m, q[d]);
  float e[32], s = 0.f;
#pragma unroll
  for (int d = 0; d < 32; ++d) { e[d] = expf(q[d] - m); s += e[d]; }
  float inv = 0.17677669529663687f / s;   // softmax * d^-0.5
#pragma unroll
  for (int d = 0; d < 32; ++d) q[d] = e[d] * inv;
}

__global__ void ksoftmax_k(const float* __restrict__ KV, const float* __restrict__ kvb,
                           float* __restrict__ KS, float* __restrict__ KSpad,
                           int nt, int padlen)
{
  int blk = blockIdx.x;
  int b = blk / 15, h = blk % 15 + 1;
  int d = threadIdx.x;                  // 0..31
  int col = h * HD + d;
  const float* src = KV + (size_t)b * nt * DIM + col;
  float pv = kvb[col];
  float m = (padlen > 0) ? pv : -1e30f;
  for (int t = 0; t < nt; ++t) m = fmaxf(m, src[(size_t)t * DIM]);
  float s = (float)padlen * expf(pv - m);
  for (int t = 0; t < nt; ++t) s += expf(src[(size_t)t * DIM] - m);
  float inv = 1.f / s;
  float* dst = KS + (size_t)b * nt * DIM + col;
  for (int t = 0; t < nt; ++t) dst[(size_t)t * DIM] = expf(src[(size_t)t * DIM] - m) * inv;
  KSpad[b * DIM + col] = expf(pv - m) * inv;
}

__global__ __launch_bounds__(256) void ctx_k(
    const float* __restrict__ KS, const float* __restrict__ KSpad,
    const float* __restrict__ V, const float* __restrict__ kvb,
    float* __restrict__ CTX, int nt, int padlen)
{
  int blk = blockIdx.x;
  int b = blk / 15, h = blk % 15 + 1;
  __shared__ float ksS[32][32];
  __shared__ float vS[32][33];
  int tid = threadIdx.x;
  float acc[4] = {0.f, 0.f, 0.f, 0.f};
  for (int t0 = 0; t0 < nt; t0 += 32) {
    int nchunk = nt - t0; if (nchunk > 32) nchunk = 32;
    for (int e = tid; e < 32 * 32; e += 256) {
      int tt = e >> 5, d = e & 31;
      float kv = 0.f, vv = 0.f;
      if (tt < nchunk) {
        size_t ofs = (size_t)(b * nt + t0 + tt) * DIM + h * HD + d;
        kv = KS[ofs]; vv = V[ofs];
      }
      ksS[tt][d] = kv; vS[tt][d] = vv;
    }
    __syncthreads();
#pragma unroll
    for (int i = 0; i < 4; ++i) {
      int pi = tid + i * 256;
      int d = pi >> 5, e = pi & 31;
      float a = acc[i];
#pragma unroll
      for (int tt = 0; tt < 32; ++tt) a += ksS[tt][d] * vS[tt][e];
      acc[i] = a;
    }
    __syncthreads();
  }
#pragma unroll
  for (int i = 0; i < 4; ++i) {
    int pi = tid + i * 256;
    int d = pi >> 5, e = pi & 31;
    float a = acc[i] + (float)padlen * KSpad[b * DIM + h * HD + d] * kvb[h * HD + e];
    CTX[(((size_t)(b * 16 + h)) * 32 + d) * 32 + e] = a;
  }
}

__global__ __launch_bounds__(256) void lin_out_k(
    const float* __restrict__ QS, const float* __restrict__ CTX,
    float* __restrict__ Out, int nt)
{
  int blk = blockIdx.x;
  int b = blk / 15, h = blk % 15 + 1;
  __shared__ float ctxS[32][32];
  for (int e = threadIdx.x; e < 1024; e += 256)
    ctxS[e >> 5][e & 31] = CTX[((size_t)(b * 16 + h)) * 1024 + e];
  __syncthreads();
  int wave = threadIdx.x >> 5, lane = threadIdx.x & 31;
  for (int t = wave; t < nt; t += 8) {
    const float* q = QS + (size_t)(b * nt + t) * DIM + h * HD;
    float s = 0.f;
#pragma unroll
    for (int d = 0; d < 32; ++d) s += q[d] * ctxS[d][lane];
    Out[(size_t)(b * nt + t) * DIM + h * HD + lane] = s;
  }
}

// --------------------------- decoder assembly -----------------------------
__global__ void dec_assemble_k(const float* __restrict__ xd, const float* __restrict__ mask_token,
                               const int* __restrict__ ids_restore, const int* __restrict__ idx,
                               const float* __restrict__ pos_embed, float* __restrict__ X)
{
  int row = blockIdx.x;            // b*DEC_N + j
  int b = row / DEC_N, j = row % DEC_N;
  int i = (j == 0) ? 0 : (j - 1);
  int r = ids_restore[b * SEQL + i];
  const float* src = (r < LEN_KEEP) ? (xd + (size_t)(b * ENC_N + 1 + r) * DIM) : mask_token;
  float* xr = X + (size_t)row * DIM;
  if (j == 0) {
    const float* pe = pos_embed + (size_t)idx[b * SEQL + 0] * DIM;
    for (int c = threadIdx.x; c < DIM; c += blockDim.x) xr[c] = src[c] + pe[c];
  } else {
    for (int c = threadIdx.x; c < DIM; c += blockDim.x) xr[c] = src[c];
  }
}

__global__ void cls_copy_k(const float* __restrict__ latent_ln, float* __restrict__ out_cls)
{
  int b = blockIdx.x;
  for (int c = threadIdx.x; c < DIM; c += blockDim.x)
    out_cls[b * DIM + c] = latent_ln[(size_t)b * ENC_N * DIM + c];
}

// ---------------------------- pred + loss ---------------------------------
__global__ __launch_bounds__(256) void pred_loss_k(
    const float* __restrict__ xln, const float* __restrict__ pw, const float* __restrict__ pb,
    const float* __restrict__ expr, const float* __restrict__ mask,
    float* __restrict__ pred_out, float* __restrict__ pnum, float* __restrict__ pden)
{
  __shared__ float snum[8], sden[8];
  int wave = threadIdx.x >> 5, lane = threadIdx.x & 31;
  int e = blockIdx.x * 8 + wave;          // 0..15999
  int b = e / SEQL, i = e % SEQL;
  const float* xr = xln + (size_t)(b * DEC_N + i + 1) * DIM;
  float s = 0.f;
  for (int c = lane; c < DIM; c += 32) s += xr[c] * pw[c];
  for (int off = 16; off; off >>= 1) s += __shfl_xor(s, off, 32);
  if (lane == 0) {
    float p = s + pb[0];
    pred_out[e] = p;
    float ex = expr[e]; ex = (ex != ex) ? 0.f : ex;   // nan_to_num
    float mk = mask[e];
    float df = p - ex;
    snum[wave] = df * df * mk;
    sden[wave] = mk;
  }
  __syncthreads();
  if (threadIdx.x == 0) {
    float n = 0.f, d = 0.f;
    for (int w = 0; w < 8; ++w) { n += snum[w]; d += sden[w]; }
    pnum[blockIdx.x] = n; pden[blockIdx.x] = d;
  }
}

__global__ void loss_fin_k(const float* __restrict__ pnum, const float* __restrict__ pden,
                           float* __restrict__ out, int nb)
{
  if (threadIdx.x == 0) {
    float n = 0.f, d = 0.f;
    for (int i = 0; i < nb; ++i) { n += pnum[i]; d += pden[i]; }
    out[0] = n / d;
  }
}

// ------------------------------ host side ---------------------------------
struct Layer {
  const float *n1w,*n1b,*qw,*qb,*kvw,*kvb,*pw,*pb,*n2w,*n2b,*f1w,*f1b,*f2w,*f2b;
};

static Layer get_layer(void* const* d_in, int base, int d) {
  Layer L;
  L.n1w = (const float*)d_in[base+0]  + (size_t)d * DIM;
  L.n1b = (const float*)d_in[base+1]  + (size_t)d * DIM;
  L.qw  = (const float*)d_in[base+2]  + (size_t)d * DIM * DIM;
  L.qb  = (const float*)d_in[base+3]  + (size_t)d * DIM;
  L.kvw = (const float*)d_in[base+4]  + (size_t)d * DIM * DIM;
  L.kvb = (const float*)d_in[base+5]  + (size_t)d * DIM;
  L.pw  = (const float*)d_in[base+6]  + (size_t)d * DIM * DIM;
  L.pb  = (const float*)d_in[base+7]  + (size_t)d * DIM;
  L.n2w = (const float*)d_in[base+8]  + (size_t)d * DIM;
  L.n2b = (const float*)d_in[base+9]  + (size_t)d * DIM;
  L.f1w = (const float*)d_in[base+10] + (size_t)d * DIM * 4 * DIM;
  L.f1b = (const float*)d_in[base+11] + (size_t)d * 4 * DIM;
  L.f2w = (const float*)d_in[base+12] + (size_t)d * 4 * DIM * DIM;
  L.f2b = (const float*)d_in[base+13] + (size_t)d * DIM;
  return L;
}

static void run_block(const Layer& L, float* buf_x, float* buf_ln, float* buf_q,
                      float* buf_kv, float* buf_ks, float* buf_kspad, float* buf_hid,
                      float* buf_ctx, int nt, hipStream_t stream)
{
  const int rows   = BATCH * nt;
  const int rem    = nt % 64;
  const int padlen = rem ? (64 - rem) : 0;
  const int npad   = nt + padlen;
  const int nw     = npad / 64;
  const int gLN    = (rows + 7) / 8;
  dim3 g512((DIM + BN - 1) / BN, (rows + BM - 1) / BM);
  dim3 g2048((4 * DIM + BN - 1) / BN, (rows + BM - 1) / BM);

  layernorm_k<<<gLN, 256, 0, stream>>>(buf_x, L.n1w, L.n1b, buf_ln, rows);
  gemm_wmma<<<g512, 256, 0, stream>>>(buf_ln, L.qw,  L.qb,  nullptr, buf_q,  rows, DIM, DIM, 0);
  gemm_wmma<<<g512, 256, 0, stream>>>(buf_ln, L.kvw, L.kvb, nullptr, buf_kv, rows, DIM, DIM, 0);
  local_attn_k<<<dim3(nw, BATCH), 256, 0, stream>>>(buf_q, buf_kv, L.kvb, buf_ln, nt, npad, padlen);
  qsoftmax_k<<<(rows * 15 + 255) / 256, 256, 0, stream>>>(buf_q, rows);
  ksoftmax_k<<<BATCH * 15, 32, 0, stream>>>(buf_kv, L.kvb, buf_ks, buf_kspad, nt, padlen);
  ctx_k<<<BATCH * 15, 256, 0, stream>>>(buf_ks, buf_kspad, buf_kv, L.kvb, buf_ctx, nt, padlen);
  lin_out_k<<<BATCH * 15, 256, 0, stream>>>(buf_q, buf_ctx, buf_ln, nt);
  gemm_wmma<<<g512, 256, 0, stream>>>(buf_ln, L.pw, L.pb, buf_x, buf_x, rows, DIM, DIM, 0);
  layernorm_k<<<gLN, 256, 0, stream>>>(buf_x, L.n2w, L.n2b, buf_ln, rows);
  gemm_wmma<<<g2048, 256, 0, stream>>>(buf_ln, L.f1w, L.f1b, nullptr, buf_hid, rows, 4 * DIM, DIM, 1);
  gemm_wmma<<<g512, 256, 0, stream>>>(buf_hid, L.f2w, L.f2b, buf_x, buf_x, rows, DIM, 4 * DIM, 0);
}

extern "C" void kernel_launch(void* const* d_in, const int* in_sizes, int n_in,
                              void* d_out, int out_size, void* d_ws, size_t ws_size,
                              hipStream_t stream) {
  (void)in_sizes; (void)n_in; (void)out_size; (void)ws_size;
  const float* expr       = (const float*)d_in[0];
  const int*   idx        = (const int*)  d_in[1];
  const float* noise      = (const float*)d_in[2];
  const float* pos_embed  = (const float*)d_in[3];
  const float* cls_token  = (const float*)d_in[4];
  const float* mask_token = (const float*)d_in[5];
  const float* eew        = (const float*)d_in[6];
  const float* eeb        = (const float*)d_in[7];
  const float* enc_nw     = (const float*)d_in[22];
  const float* enc_nb     = (const float*)d_in[23];
  const float* dew        = (const float*)d_in[24];
  const float* deb        = (const float*)d_in[25];
  const float* dec_nw     = (const float*)d_in[40];
  const float* dec_nb     = (const float*)d_in[41];
  const float* pred_w     = (const float*)d_in[42];
  const float* pred_b     = (const float*)d_in[43];

  float* out      = (float*)d_out;
  float* out_pred = out + 1;
  float* out_mask = out + 1 + BATCH * SEQL;
  float* out_cls  = out + 1 + 2 * BATCH * SEQL;

  // ---- workspace layout ----
  float* ws       = (float*)d_ws;
  float* buf_x    = ws;
  float* buf_ln   = buf_x  + (size_t)WSROWS * DIM;
  float* buf_q    = buf_ln + (size_t)WSROWS * DIM;
  float* buf_kv   = buf_q  + (size_t)WSROWS * DIM;
  float* buf_ks   = buf_kv + (size_t)WSROWS * DIM;
  float* buf_hid  = buf_ks + (size_t)WSROWS * DIM;                 // WSROWS*2048
  float* buf_ctx  = buf_hid + (size_t)WSROWS * 4 * DIM;            // 8*16*32*32
  float* pnum     = buf_ctx + (size_t)BATCH * 16 * 32 * 32;        // 2048
  float* pden     = pnum + 2048;                                   // 2048
  float* buf_kspad= buf_ks + (size_t)(BATCH * DEC_N) * DIM;        // tail rows of buf_ks
  int*   ids_shuffle = (int*)(pden + 2048);
  int*   ids_restore = ids_shuffle + BATCH * SEQL;

  // 1) masking: stable argsort ranks, ids, and mask output
  argsort_k<<<BATCH, 256, 0, stream>>>(noise, ids_shuffle, ids_restore, out_mask);

  // 2) encoder input embed + gather (cls + kept tokens)
  enc_embed_k<<<BATCH * ENC_N, 256, 0, stream>>>(expr, idx, ids_shuffle, pos_embed,
                                                 cls_token, eew, eeb, buf_x);

  // 3) encoder blocks
  for (int d = 0; d < ENC_DEPTH; ++d) {
    Layer L = get_layer(d_in, 8, d);
    run_block(L, buf_x, buf_ln, buf_q, buf_kv, buf_ks, buf_kspad, buf_hid, buf_ctx, ENC_N, stream);
  }
  // 4) final encoder LN -> latent; copy cls latent to output
  {
    int rows = BATCH * ENC_N;
    layernorm_k<<<(rows + 7) / 8, 256, 0, stream>>>(buf_x, enc_nw, enc_nb, buf_ln, rows);
    cls_copy_k<<<BATCH, 256, 0, stream>>>(buf_ln, out_cls);
    // 5) decoder embed: xd = latent @ dec_embed_w + b
    dim3 g((DIM + BN - 1) / BN, (rows + BM - 1) / BM);
    gemm_wmma<<<g, 256, 0, stream>>>(buf_ln, dew, deb, nullptr, buf_q, rows, DIM, DIM, 0);
  }
  // 6) decoder sequence assembly (unshuffle + mask tokens + pos-embed quirk)
  dec_assemble_k<<<BATCH * DEC_N, 256, 0, stream>>>(buf_q, mask_token, ids_restore,
                                                    idx, pos_embed, buf_x);
  // 7) decoder blocks
  for (int d = 0; d < DEC_DEPTH; ++d) {
    Layer L = get_layer(d_in, 26, d);
    run_block(L, buf_x, buf_ln, buf_q, buf_kv, buf_ks, buf_kspad, buf_hid, buf_ctx, DEC_N, stream);
  }
  // 8) final decoder LN
  {
    int rows = BATCH * DEC_N;
    layernorm_k<<<(rows + 7) / 8, 256, 0, stream>>>(buf_x, dec_nw, dec_nb, buf_ln, rows);
  }
  // 9) prediction head + deterministic masked-MSE loss
  const int nblk = (BATCH * SEQL) / 8;   // 2000
  pred_loss_k<<<nblk, 256, 0, stream>>>(buf_ln, pred_w, pred_b, expr, out_mask,
                                        out_pred, pnum, pden);
  loss_fin_k<<<1, 32, 0, stream>>>(pnum, pden, out, nblk);
}
